// WaveNet_22170621182656
// MI455X (gfx1250) — matrix operand order repaired and checked by reading, not compiled, not run on hardware
//
#include <hip/hip_runtime.h>
#include <hip/hip_bf16.h>

typedef __attribute__((ext_vector_type(16))) __bf16 v16bf;
typedef __attribute__((ext_vector_type(8)))  float  v8f;

#define NLAYER 30
#define TSTEPS 128

// dilations: 3 blocks of 1..512
__device__ const int d_dil[NLAYER] = {
    1,2,4,8,16,32,64,128,256,512,
    1,2,4,8,16,32,64,128,256,512,
    1,2,4,8,16,32,64,128,256,512};
// queue slot base (only layers with d<128 get storage; others never touch the queue for T=128)
__device__ const int d_qbase[NLAYER] = {
    0,1,3,7,15,31,63,-1,-1,-1,
    127,128,130,134,142,158,190,-1,-1,-1,
    254,255,257,261,269,285,317,-1,-1,-1};

// workspace layout (bf16 element offsets)
#define WK_OFF 0            // 30*2*8*2*512  = 491520   conv kernels
#define WR_OFF 491520       // 30*4*2*512    = 122880   res_w
#define WS_OFF 614400       // 30*16*2*512   = 491520   skip_w
#define W0_OFF 1105920      // 16*8*512      = 65536    out_w0
#define W1_OFF 1171456      // 16*8*512      = 65536    out_w1
#define Q_OFF  1236992      // 381*8*1024    = 3121152  dilation queues (A-layout tiles)
#define WS_TOTAL 4358144    // bf16 elems  (~8.7 MB)

static __device__ __forceinline__ v8f wmma_bf16(v16bf a, v16bf b, v8f c) {
    return __builtin_amdgcn_wmma_f32_16x16x32_bf16(false, a, false, b, (short)0, c, false, false);
}

// fast sigmoid via v_exp_f32 (__expf) + v_rcp_f32
static __device__ __forceinline__ float fast_sigmoid(float x) {
    return __builtin_amdgcn_rcpf(1.f + __expf(-x));
}

// Convert f32 weight matrices (K x N, row-major) into WMMA B-register layout:
// dst[mat][nt][kc][lane][e] = W[kc*32 + (e<8 ? e : e+8) + 8*(lane>>4)][nt*16 + (lane&15)]
__global__ void prep_weights(const float* __restrict__ src, __bf16* __restrict__ dst,
                             int nmat, int K, int N) {
    int ntk = K >> 5, ntn = N >> 4;
    long msz = (long)ntn * ntk * 512;
    long total = (long)nmat * msz;
    long f = (long)blockIdx.x * blockDim.x + threadIdx.x;
    if (f >= total) return;
    int mat = (int)(f / msz);
    int r   = (int)(f % msz);
    int nt = r / (ntk * 512);
    int r2 = r % (ntk * 512);
    int kc = r2 >> 9;
    int r3 = r2 & 511;
    int lane = r3 >> 4, e = r3 & 15;
    int n = nt * 16 + (lane & 15);
    int k = kc * 32 + (e < 8 ? e : e + 8) + 8 * (lane >> 4);
    dst[f] = (__bf16)src[(long)mat * K * N + (long)k * N + n];
}

__global__ __launch_bounds__(256)
void wavenet_main(const int* __restrict__ seed,
                  const float* __restrict__ emb,
                  const float* __restrict__ conv_bias,
                  const float* __restrict__ res_b,
                  const float* __restrict__ skip_b,
                  const float* __restrict__ out_b0,
                  const float* __restrict__ out_b1,
                  __bf16* __restrict__ ws,
                  int* __restrict__ out_samples,
                  float* __restrict__ out_logits) {
    const int wg   = blockIdx.x;          // 0..7, owns batch rows [wg*16, wg*16+16)
    const int tid  = threadIdx.x;         // 0..255
    const int lane = tid & 31;
    const int wave = tid >> 5;            // 0..7
    const int ln16 = lane & 15;
    const int lh   = lane >> 4;

    __shared__ float xbuf[16 * 65];       // residual activation x (16 x 64, padded)
    __shared__ float hbuf[16 * 129];      // conv pre-activation (16 x 128, padded)
    __shared__ float skipbuf[16 * 257];   // skip accumulator (16 x 256, padded)
    __shared__ float h0buf[16 * 257];     // head hidden, reused for logits
    __shared__ __align__(32) __bf16 aX[1024];   // x in A-layout (2 kchunks)
    __shared__ __align__(32) __bf16 aL[1024];   // x_last in A-layout
    __shared__ __align__(32) __bf16 aG[1024];   // gated act in A-layout
    __shared__ __align__(32) __bf16 aBig[4096]; // K=256 A-layout (skip, then h0)
    __shared__ float redV[256];
    __shared__ int   redI[256];
    __shared__ int   nxt[16];

    const __bf16* WK  = ws + WK_OFF;
    const __bf16* WR  = ws + WR_OFF;
    const __bf16* WSk = ws + WS_OFF;
    const __bf16* W0  = ws + W0_OFF;
    const __bf16* W1  = ws + W1_OFF;
    __bf16*       Q   = ws + Q_OFF;

    if (tid < 16) nxt[tid] = seed[wg * 16 + tid];
    __syncthreads();

    for (int t = 0; t < TSTEPS; ++t) {
        // ---- step start: x = embedding[nxt]; skip = 0 ----
        for (int u = tid; u < 16 * 64; u += 256) {
            int m = u >> 6, c = u & 63;
            xbuf[m * 65 + c] = emb[nxt[m] * 64 + c];
        }
        for (int u = tid; u < 16 * 256; u += 256) {
            int m = u >> 8, n = u & 255;
            skipbuf[m * 257 + n] = 0.f;
        }

        for (int i = 0; i < NLAYER; ++i) {
            const int d  = d_dil[i];
            const int qb = d_qbase[i];
            __syncthreads();  // xbuf/skipbuf writes from previous phase visible

            // ---- queue pop/push + build aX / aL (A-register layout) ----
            {
                const bool doRead  = (qb >= 0) && (t >= d);
                const bool doWrite = (qb >= 0) && (t + d < TSTEPS);
                int qoff = 0;
                if (qb >= 0) qoff = ((qb + (t % d)) * 8 + wg) * 1024;
                for (int u = tid; u < 1024; u += 256) {
                    int l = (u >> 4) & 31, e = u & 15;
                    int m = l & 15, hh = l >> 4;
                    int k = ((u >> 9) << 5) + (e < 8 ? e : e + 8) + 8 * hh; // A layout
                    __bf16 old;
                    if (doRead) old = Q[qoff + u]; else old = (__bf16)0.0f;
                    aL[u] = old;
                    __bf16 nv = (__bf16)xbuf[m * 65 + k];
                    aX[u] = nv;
                    if (doWrite) Q[qoff + u] = nv;
                }
            }
            __syncthreads();

            // ---- conv: h = x_last@k0 + x@k1 + bias ; wave w owns N-tile w ----
            {
                const int nt = wave;
                v8f acc;
                float bv = conv_bias[i * 128 + nt * 16 + ln16];
#pragma unroll
                for (int r = 0; r < 8; ++r) acc[r] = bv;
                const v16bf* aXv = (const v16bf*)aX;
                const v16bf* aLv = (const v16bf*)aL;
                const v16bf* wk0 = (const v16bf*)(WK + (long)(((i * 2 + 0) * 8 + nt) * 2) * 512);
                const v16bf* wk1 = (const v16bf*)(WK + (long)(((i * 2 + 1) * 8 + nt) * 2) * 512);
                if (t >= d) {  // x_last is all-zero otherwise: skip its WMMAs
#pragma unroll
                    for (int kc = 0; kc < 2; ++kc)
                        acc = wmma_bf16(aLv[kc * 32 + lane], wk0[kc * 32 + lane], acc);
                }
#pragma unroll
                for (int kc = 0; kc < 2; ++kc)
                    acc = wmma_bf16(aXv[kc * 32 + lane], wk1[kc * 32 + lane], acc);
                int n = nt * 16 + ln16, mh = lh * 8;
#pragma unroll
                for (int r = 0; r < 8; ++r) hbuf[(mh + r) * 129 + n] = acc[r];
            }
            __syncthreads();

            // ---- gate: g = tanh(h[:, :64]) * sigmoid(h[:, 64:]) -> aG (A layout) ----
            // tanh(x) = 2*sigmoid(2x) - 1; both via v_exp_f32 + v_rcp_f32 (TRANS ops)
            for (int u = tid; u < 1024; u += 256) {
                int l = (u >> 4) & 31, e = u & 15;
                int m = l & 15, hh = l >> 4;
                int k = ((u >> 9) << 5) + (e < 8 ? e : e + 8) + 8 * hh;
                float h1 = hbuf[m * 129 + k];
                float h2 = hbuf[m * 129 + k + 64];
                float th = 2.f * fast_sigmoid(2.f * h1) - 1.f;
                float g  = th * fast_sigmoid(h2);
                aG[u] = (__bf16)g;
            }
            __syncthreads();

            // ---- res (4 tiles) + skip (16 tiles): 20 jobs over 8 waves ----
            {
                const v16bf* aGv = (const v16bf*)aG;
                v16bf a0 = aGv[lane];
                v16bf a1 = aGv[32 + lane];
                for (int job = wave; job < 20; job += 8) {
                    if (job < 4) {
                        int nt = job;
                        v8f acc;
                        float bv = res_b[i * 64 + nt * 16 + ln16];
#pragma unroll
                        for (int r = 0; r < 8; ++r) acc[r] = bv;
                        const v16bf* w = (const v16bf*)(WR + (long)((i * 4 + nt) * 2) * 512);
                        acc = wmma_bf16(a0, w[lane], acc);
                        acc = wmma_bf16(a1, w[32 + lane], acc);
                        int n = nt * 16 + ln16, mh = lh * 8;
#pragma unroll
                        for (int r = 0; r < 8; ++r) {
                            int idx = (mh + r) * 65 + n;
                            xbuf[idx] = xbuf[idx] + acc[r];
                        }
                    } else {
                        int nt = job - 4;
                        v8f acc;
                        float bv = skip_b[i * 256 + nt * 16 + ln16];
#pragma unroll
                        for (int r = 0; r < 8; ++r) acc[r] = bv;
                        const v16bf* w = (const v16bf*)(WSk + (long)((i * 16 + nt) * 2) * 512);
                        acc = wmma_bf16(a0, w[lane], acc);
                        acc = wmma_bf16(a1, w[32 + lane], acc);
                        int n = nt * 16 + ln16, mh = lh * 8;
#pragma unroll
                        for (int r = 0; r < 8; ++r) {
                            int idx = (mh + r) * 257 + n;
                            skipbuf[idx] += acc[r];
                        }
                    }
                }
            }
        } // layers

        __syncthreads();
        // ---- head: aS = bf16(relu(skip)), K=256 A layout ----
        for (int u = tid; u < 4096; u += 256) {
            int l = (u >> 4) & 31, e = u & 15;
            int m = l & 15, hh = l >> 4;
            int k = ((u >> 9) << 5) + (e < 8 ? e : e + 8) + 8 * hh;
            float s = skipbuf[m * 257 + k];
            aBig[u] = (__bf16)(s > 0.f ? s : 0.f);
        }
        __syncthreads();
        // ---- h0 = relu(skip @ out_w0 + b0) ----
        {
            const v16bf* av = (const v16bf*)aBig;
            for (int nt = wave; nt < 16; nt += 8) {
                v8f acc;
                float bv = out_b0[nt * 16 + ln16];
#pragma unroll
                for (int r = 0; r < 8; ++r) acc[r] = bv;
                const v16bf* w = (const v16bf*)(W0 + (long)(nt * 8) * 512);
#pragma unroll
                for (int kc = 0; kc < 8; ++kc)
                    acc = wmma_bf16(av[kc * 32 + lane], w[kc * 32 + lane], acc);
                int n = nt * 16 + ln16, mh = lh * 8;
#pragma unroll
                for (int r = 0; r < 8; ++r) {
                    float v = acc[r];
                    h0buf[(mh + r) * 257 + n] = v > 0.f ? v : 0.f;
                }
            }
        }
        __syncthreads();
        // ---- aH = bf16(h0) ----
        for (int u = tid; u < 4096; u += 256) {
            int l = (u >> 4) & 31, e = u & 15;
            int m = l & 15, hh = l >> 4;
            int k = ((u >> 9) << 5) + (e < 8 ? e : e + 8) + 8 * hh;
            aBig[u] = (__bf16)h0buf[m * 257 + k];
        }
        __syncthreads();
        // ---- logits = h0 @ out_w1 + b1 (reuse h0buf; also write to global) ----
        {
            const v16bf* av = (const v16bf*)aBig;
            for (int nt = wave; nt < 16; nt += 8) {
                v8f acc;
                float bv = out_b1[nt * 16 + ln16];
#pragma unroll
                for (int r = 0; r < 8; ++r) acc[r] = bv;
                const v16bf* w = (const v16bf*)(W1 + (long)(nt * 8) * 512);
#pragma unroll
                for (int kc = 0; kc < 8; ++kc)
                    acc = wmma_bf16(av[kc * 32 + lane], w[kc * 32 + lane], acc);
                int n = nt * 16 + ln16, mh = lh * 8;
#pragma unroll
                for (int r = 0; r < 8; ++r) {
                    float v = acc[r];
                    h0buf[(mh + r) * 257 + n] = v;
                    out_logits[((long)(wg * 16 + mh + r) * TSTEPS + t) * 256 + n] = v;
                }
            }
        }
        __syncthreads();
        // ---- argmax (first-max tie-break): 16 threads/row, 16 elems each ----
        {
            int m = tid >> 4, j = tid & 15;
            float best = -3.4e38f; int bi = 0;
            for (int q = 0; q < 16; ++q) {
                int v = j * 16 + q;
                float x = h0buf[m * 257 + v];
                if (x > best) { best = x; bi = v; }
            }
            redV[m * 16 + j] = best;
            redI[m * 16 + j] = bi;
        }
        __syncthreads();
        if (tid < 16) {
            int m = tid; float best = -3.4e38f; int bi = 0;
            for (int j = 0; j < 16; ++j) {
                float v = redV[m * 16 + j];
                if (v > best) { best = v; bi = redI[m * 16 + j]; }
            }
            nxt[m] = bi;
            out_samples[(wg * 16 + m) * TSTEPS + t] = bi;
        }
        __syncthreads();
    } // t
}

extern "C" void kernel_launch(void* const* d_in, const int* in_sizes, int n_in,
                              void* d_out, int out_size, void* d_ws, size_t ws_size,
                              hipStream_t stream) {
    const int*   seed      = (const int*)d_in[0];
    const float* embedding = (const float*)d_in[1];
    const float* kernels   = (const float*)d_in[2];
    const float* conv_bias = (const float*)d_in[3];
    const float* res_w     = (const float*)d_in[4];
    const float* res_b     = (const float*)d_in[5];
    const float* skip_w    = (const float*)d_in[6];
    const float* skip_b    = (const float*)d_in[7];
    const float* out_w0    = (const float*)d_in[8];
    const float* out_b0    = (const float*)d_in[9];
    const float* out_w1    = (const float*)d_in[10];
    const float* out_b1    = (const float*)d_in[11];
    (void)in_sizes; (void)n_in; (void)out_size; (void)ws_size;

    __bf16* ws = (__bf16*)d_ws;
    int*   out_samples = (int*)d_out;
    float* out_logits  = (float*)d_out + 128 * 128;

    // pre-convert weights to bf16 WMMA B-register layout in workspace
    auto nb = [](long total) { return (int)((total + 255) / 256); };
    prep_weights<<<nb(60L * 8 * 2 * 512), 256, 0, stream>>>(kernels, ws + WK_OFF, 60, 64, 128);
    prep_weights<<<nb(30L * 4 * 2 * 512), 256, 0, stream>>>(res_w,   ws + WR_OFF, 30, 64, 64);
    prep_weights<<<nb(30L * 16 * 2 * 512), 256, 0, stream>>>(skip_w, ws + WS_OFF, 30, 64, 256);
    prep_weights<<<nb(16L * 8 * 512), 256, 0, stream>>>(out_w0, ws + W0_OFF, 1, 256, 256);
    prep_weights<<<nb(16L * 8 * 512), 256, 0, stream>>>(out_w1, ws + W1_OFF, 1, 256, 256);

    // persistent generation: 8 independent workgroups, one 16-row batch slice each
    wavenet_main<<<8, 256, 0, stream>>>(seed, embedding, conv_bias, res_b, skip_b,
                                        out_b0, out_b1, ws, out_samples, out_logits);
}
